// SAGE_83270825935423
// MI455X (gfx1250) — matrix operand order repaired and checked by reading, not compiled, not run on hardware
//
#include <hip/hip_runtime.h>
#include <hip/hip_bf16.h>
#include <math.h>

// ---------------------------------------------------------------------------
// GraphSAGE (pool aggregator), 2 layers, for MI455X (gfx1250, wave32, WMMA).
//  - GEMMs: bf16 inputs, fp32 accumulate via v_wmma_f32_16x16x32_bf16;
//    each wave computes a 16x64 strip (4 WMMAs per K-step share one A frag)
//  - fragment loads are pure global_load_b128 with pointer increments
//  - segment_max: edge-parallel global_atomic_max_i32 (valid: pooled >= 0)
//  - all node matrices are L2-resident (192 MB L2)
// ---------------------------------------------------------------------------

typedef __attribute__((ext_vector_type(16))) __bf16 v16bf;
typedef __attribute__((ext_vector_type(8)))  float  v8f;

#define N_NODES 50000
#define N_EDGES 800000
#define IN_F    128
#define HID_F   256
#define OUT_F   128

// ---------------- weight packing: W[Dout,K] f32 -> B-fragment bf16 tiles ----
// Tile (nt, kt) covers B rows k = kt*32..+31, cols n = nt*16..+15 where
// B[k][n] = W[n][k].  Per ISA 7.12.2 (16-bit B, 32x16):
//   lanes 0-15 hold K=0..7,16..23 of the tile (col n = lane), lanes 16-31 hold
//   K=8..15,24..31 (col n = lane-16); VGPR v packs {k, k+1}, k = 16*(v>>2)+8*(lane>>4)+2*(v&3), in {lo16, hi16}.
// Stored so lane l reads 8 consecutive dwords: tile*256 + l*8 + v.
__global__ void pack_w_kernel(const float* __restrict__ W,
                              unsigned int* __restrict__ Wp,
                              int Dout, int K) {
    int t = blockIdx.x * blockDim.x + threadIdx.x;   // one dword per thread
    int kTiles = K >> 5;
    int total  = (Dout >> 4) * kTiles * 256;
    if (t >= total) return;
    int tile = t >> 8;
    int idx  = t & 255;
    int lane = idx >> 3;
    int v    = idx & 7;
    int nt = tile / kTiles;
    int kt = tile % kTiles;
    int n  = nt * 16 + (lane & 15);
    int k0 = kt * 32 + (v >> 2) * 16 + (lane >> 4) * 8 + 2 * (v & 3);
    union { __bf16 b[2]; unsigned int u; } pk;
    pk.b[0] = (__bf16)W[(size_t)n * K + k0];
    pk.b[1] = (__bf16)W[(size_t)n * K + k0 + 1];
    Wp[t] = pk.u;
}

// ---------------- f32 -> bf16 elementwise ----------------------------------
__global__ void f32_to_bf16_kernel(const float* __restrict__ in,
                                   __bf16* __restrict__ out, int n) {
    int i = blockIdx.x * blockDim.x + threadIdx.x;
    if (i < n) out[i] = (__bf16)in[i];
}

// ---------------- WMMA GEMM accumulation core ------------------------------
// A fragment (16x32 bf16, ISA 7.12.2): per lane, two contiguous 16B runs:
//   lanes 0-15 : K=0..7  (VGPR 0-3) and K=16..23 (VGPR 4-7), row m = lane
//   lanes 16-31: K=8..15 and K=24..31, row m = lane-16
// -> exactly two global_load_b128 per fragment, advancing 64 B per K-step.
// B fragment: 8 consecutive dwords per lane in the packed tile (two b128s),
// consecutive kt tiles are 1 KB apart.
__device__ __forceinline__ void gemm_accum(const __bf16* __restrict__ A,
                                           const unsigned int* __restrict__ Wp,
                                           int K, int m0, int nt0, int lane,
                                           v8f c[4]) {
    const int hh = lane >> 4;
    const int kTiles = K >> 5;
    const uint4* ap = reinterpret_cast<const uint4*>(
        A + (size_t)(m0 + (lane & 15)) * K + (hh ? 8 : 0));
    const uint4* bp = reinterpret_cast<const uint4*>(Wp) +
                      (size_t)nt0 * kTiles * 64 + lane * 2;
    const size_t bsub = (size_t)kTiles * 64;     // uint4 stride between n-subtiles
    for (int kt = 0; kt < kTiles; ++kt) {
        union { v16bf v; uint4 q[2]; } a;
        a.q[0] = ap[0];
        a.q[1] = ap[2];                          // +16 elements = +32 B
        ap += 4;                                 // next K-step: +64 B
#pragma unroll
        for (int j = 0; j < 4; ++j) {
            union { v16bf v; uint4 q[2]; } b;
            b.q[0] = bp[j * bsub];
            b.q[1] = bp[j * bsub + 1];
            c[j] = __builtin_amdgcn_wmma_f32_16x16x32_bf16(
                false, a.v, false, b.v, (short)0, c[j], false, false);
        }
        bp += 64;                                // next K-step: +1 KB
    }
}

// ---------------- dual-input WMMA GEMM -------------------------------------
// OUT[M,DN] = A1[M,K1]@W1p(+ A2[M,K2]@W2p) + bias, optional ReLU.
// 8 waves/block; each wave owns a 16x64 output strip (4 accumulators).
// EXEC stays all-ones for any wave reaching WMMA (uniform early-out per wave).
__global__ __launch_bounds__(256) void wmma_dual_gemm(
    const __bf16* __restrict__ A1, const unsigned int* __restrict__ W1p, int K1,
    const __bf16* __restrict__ A2, const unsigned int* __restrict__ W2p, int K2,
    const float* __restrict__ bias, float* __restrict__ OUT,
    int M, int DN, int do_relu) {
    const int lane  = threadIdx.x & 31;
    const int wave  = threadIdx.x >> 5;
    const int mTile = blockIdx.x * 8 + wave;
    if (mTile * 16 >= M) return;                 // uniform per wave
    const int m0  = mTile * 16;
    const int nt0 = blockIdx.y * 4;

    v8f c[4] = {};
    gemm_accum(A1, W1p, K1, m0, nt0, lane, c);
    if (A2) gemm_accum(A2, W2p, K2, m0, nt0, lane, c);

    // C/D layout: lanes 0-15 -> rows m0..m0+7 (vgpr r = row), col = lane&15;
    //             lanes 16-31 -> rows m0+8..m0+15.
    const int ncol = lane & 15;
    float* orow = OUT + (size_t)(m0 + (lane >> 4) * 8) * DN;
#pragma unroll
    for (int j = 0; j < 4; ++j) {
        const int n  = (nt0 + j) * 16 + ncol;
        const float bb = bias ? bias[n] : 0.0f;
        float* o = orow + n;
#pragma unroll
        for (int r = 0; r < 8; ++r) {
            float v = c[j][r] + bb;
            if (do_relu) v = fmaxf(v, 0.0f);
            o[(size_t)r * DN] = v;
        }
    }
}

// ---------------- edge-parallel segment max --------------------------------
// pooled >= 0 (post-ReLU) so int ordering == float ordering; neigh pre-zeroed
// which also realizes jnp.maximum(neigh, 0) for empty segments.
__global__ void scatter_max_kernel(const float* __restrict__ pooled,
                                   const int* __restrict__ src,
                                   const int* __restrict__ dst,
                                   int* __restrict__ neigh,
                                   int logF4, int total) {
    int t = blockIdx.x * blockDim.x + threadIdx.x;
    if (t >= total) return;
    int e = t >> logF4;                          // edge
    int c = (t - (e << logF4)) * 4;              // feature chunk
    int F = 4 << logF4;
    int s = src[e];
    int d = dst[e];
    const float4 v = *reinterpret_cast<const float4*>(pooled + (size_t)s * F + c);
    int* nb = neigh + (size_t)d * F + c;
    atomicMax(nb + 0, __float_as_int(v.x));
    atomicMax(nb + 1, __float_as_int(v.y));
    atomicMax(nb + 2, __float_as_int(v.z));
    atomicMax(nb + 3, __float_as_int(v.w));
}

// ---------------- row-wise L2 normalize + ReLU -----------------------------
__global__ void l2norm_relu_kernel(const float* __restrict__ in,
                                   float* __restrict__ out_f32,
                                   __bf16* __restrict__ out_bf16, int F) {
    __shared__ float sdata[256];
    const int row = blockIdx.x, tid = threadIdx.x;
    const size_t idx = (size_t)row * F + tid;
    float v = in[idx];
    sdata[tid] = v * v;
    __syncthreads();
    for (int s = F >> 1; s > 0; s >>= 1) {
        if (tid < s) sdata[tid] += sdata[tid + s];
        __syncthreads();
    }
    float denom = fmaxf(sqrtf(sdata[0]), 1e-12f);
    float r = fmaxf(v / denom, 0.0f);
    if (out_f32)  out_f32[idx]  = r;
    if (out_bf16) out_bf16[idx] = (__bf16)r;
}

// ---------------------------------------------------------------------------
extern "C" void kernel_launch(void* const* d_in, const int* in_sizes, int n_in,
                              void* d_out, int out_size, void* d_ws, size_t ws_size,
                              hipStream_t stream) {
    const float* x   = (const float*)d_in[0];
    const int*   src = (const int*)d_in[1];
    const int*   dst = (const int*)d_in[2];
    const float* Wp1 = (const float*)d_in[3];
    const float* bp1 = (const float*)d_in[4];
    const float* Ws1 = (const float*)d_in[5];
    const float* Wn1 = (const float*)d_in[6];
    const float* b1  = (const float*)d_in[7];
    const float* Wp2 = (const float*)d_in[8];
    const float* bp2 = (const float*)d_in[9];
    const float* Ws2 = (const float*)d_in[10];
    const float* Wn2 = (const float*)d_in[11];
    const float* b2  = (const float*)d_in[12];

    char* ws = (char*)d_ws;
    size_t off = 0;
    auto alloc = [&](size_t bytes) {
        size_t o = off;
        off += (bytes + 255) & ~(size_t)255;
        return o;
    };

    const size_t oWp1p = alloc((size_t)IN_F  * IN_F  * 2);
    const size_t oWs1p = alloc((size_t)HID_F * IN_F  * 2);
    const size_t oWn1p = alloc((size_t)HID_F * IN_F  * 2);
    const size_t oWp2p = alloc((size_t)HID_F * HID_F * 2);
    const size_t oWs2p = alloc((size_t)OUT_F * HID_F * 2);
    const size_t oWn2p = alloc((size_t)OUT_F * HID_F * 2);
    const size_t oXB   = alloc((size_t)N_NODES * IN_F  * 2);
    const size_t oP1   = alloc((size_t)N_NODES * IN_F  * 4);  // pooled1
    const size_t oNG1  = alloc((size_t)N_NODES * IN_F  * 4);  // neigh1
    const size_t oN1B  = alloc((size_t)N_NODES * IN_F  * 2);
    const size_t oH1   = alloc((size_t)N_NODES * HID_F * 4);  // h1
    const size_t oH1B  = alloc((size_t)N_NODES * HID_F * 2);
    const size_t oN2B  = alloc((size_t)N_NODES * HID_F * 2);
    const size_t oO2   = alloc((size_t)N_NODES * OUT_F * 4);
    // reuse: pooled2 spans pooled1+neigh1 (both 25.6MB, contiguous);
    //        neigh2 reuses h1 (both 51.2MB, h1 dead after l2norm->h1b).
    const size_t oP2  = oP1;
    const size_t oNG2 = oH1;
    (void)ws_size; (void)in_sizes; (void)n_in; (void)out_size;

    __bf16*       xb   = (__bf16*)(ws + oXB);
    float*        p1   = (float*)(ws + oP1);
    int*          ng1  = (int*)(ws + oNG1);
    __bf16*       n1b  = (__bf16*)(ws + oN1B);
    float*        h1   = (float*)(ws + oH1);
    __bf16*       h1b  = (__bf16*)(ws + oH1B);
    float*        p2   = (float*)(ws + oP2);
    int*          ng2  = (int*)(ws + oNG2);
    __bf16*       n2b  = (__bf16*)(ws + oN2B);
    float*        o2   = (float*)(ws + oO2);
    unsigned int* wp1p = (unsigned int*)(ws + oWp1p);
    unsigned int* ws1p = (unsigned int*)(ws + oWs1p);
    unsigned int* wn1p = (unsigned int*)(ws + oWn1p);
    unsigned int* wp2p = (unsigned int*)(ws + oWp2p);
    unsigned int* ws2p = (unsigned int*)(ws + oWs2p);
    unsigned int* wn2p = (unsigned int*)(ws + oWn2p);

    const int mBlocks = ((N_NODES / 16) + 7) / 8;   // 3125 wave-tiles -> 391

    // --- pack weights to WMMA B-fragment layout ---
    pack_w_kernel<<<(IN_F/16)*(IN_F/32),   256, 0, stream>>>(Wp1, wp1p, IN_F,  IN_F);
    pack_w_kernel<<<(HID_F/16)*(IN_F/32),  256, 0, stream>>>(Ws1, ws1p, HID_F, IN_F);
    pack_w_kernel<<<(HID_F/16)*(IN_F/32),  256, 0, stream>>>(Wn1, wn1p, HID_F, IN_F);
    pack_w_kernel<<<(HID_F/16)*(HID_F/32), 256, 0, stream>>>(Wp2, wp2p, HID_F, HID_F);
    pack_w_kernel<<<(OUT_F/16)*(HID_F/32), 256, 0, stream>>>(Ws2, ws2p, OUT_F, HID_F);
    pack_w_kernel<<<(OUT_F/16)*(HID_F/32), 256, 0, stream>>>(Wn2, wn2p, OUT_F, HID_F);

    // --- layer 1 ---
    int nX = N_NODES * IN_F;
    f32_to_bf16_kernel<<<(nX + 255) / 256, 256, 0, stream>>>(x, xb, nX);

    wmma_dual_gemm<<<dim3(mBlocks, IN_F / 64), 256, 0, stream>>>(
        xb, wp1p, IN_F, nullptr, nullptr, 0, bp1, p1, N_NODES, IN_F, 1);

    hipMemsetAsync(ng1, 0, (size_t)N_NODES * IN_F * 4, stream);
    int tot1 = N_EDGES * (IN_F / 4);
    scatter_max_kernel<<<(tot1 + 255) / 256, 256, 0, stream>>>(
        p1, src, dst, ng1, 5 /* log2(128/4) */, tot1);

    f32_to_bf16_kernel<<<(nX + 255) / 256, 256, 0, stream>>>((const float*)ng1, n1b, nX);

    wmma_dual_gemm<<<dim3(mBlocks, HID_F / 64), 256, 0, stream>>>(
        xb, ws1p, IN_F, n1b, wn1p, IN_F, b1, h1, N_NODES, HID_F, 0);

    l2norm_relu_kernel<<<N_NODES, HID_F, 0, stream>>>(h1, nullptr, h1b, HID_F);

    // --- layer 2 ---
    wmma_dual_gemm<<<dim3(mBlocks, HID_F / 64), 256, 0, stream>>>(
        h1b, wp2p, HID_F, nullptr, nullptr, 0, bp2, p2, N_NODES, HID_F, 1);

    hipMemsetAsync(ng2, 0, (size_t)N_NODES * HID_F * 4, stream);
    int tot2 = N_EDGES * (HID_F / 4);
    scatter_max_kernel<<<(tot2 + 255) / 256, 256, 0, stream>>>(
        p2, src, dst, ng2, 6 /* log2(256/4) */, tot2);

    int nH = N_NODES * HID_F;
    f32_to_bf16_kernel<<<(nH + 255) / 256, 256, 0, stream>>>((const float*)ng2, n2b, nH);

    wmma_dual_gemm<<<dim3(mBlocks, OUT_F / 64), 256, 0, stream>>>(
        h1b, ws2p, HID_F, n2b, wn2p, HID_F, b2, o2, N_NODES, OUT_F, 0);

    l2norm_relu_kernel<<<N_NODES, OUT_F, 0, stream>>>(o2, (float*)d_out, nullptr, OUT_F);
}
